// PerformerAttention_80874234183844
// MI455X (gfx1250) — matrix operand order repaired and checked by reading, not compiled, not run on hardware
//
#include <hip/hip_runtime.h>
#include <hip/hip_bf16.h>
#include <stdint.h>
#include <stddef.h>

typedef __bf16 bf16;
typedef __attribute__((ext_vector_type(16))) __bf16 v16bf;
typedef __attribute__((ext_vector_type(8)))  float   v8f;

#define BN_TOT    32768     // B*N
#define C_DIM     512
#define C3_DIM    1536
#define H_NUM     8
#define D_HEAD    64
#define F_DIM     128
#define KV_CHUNKS 64

// ---------------- device helpers ----------------

// A fragment (16x32 bf16, row per lane): lane l -> row (l&15),
// K bytes: two 16B chunks at K = off and off+16 where off = 8*(l>=16).
__device__ __forceinline__ v16bf load_frag_a(const bf16* p0, int ld, int lane) {
  const bf16* p = p0 + (size_t)(lane & 15) * ld + ((lane >> 4) << 3);
  v16bf a;
  ((float4*)&a)[0] = *(const float4*)(p);
  ((float4*)&a)[1] = *(const float4*)(p + 16);
  return a;
}

// B fragment (32x16 bf16, column per lane): lane l -> column (l&15)
// = row of W; two 16B chunks at K = off and off+16 where off = 8*(l>=16).
__device__ __forceinline__ v16bf load_frag_b(const bf16* p0, int ld, int lane) {
  const bf16* p = p0 + (size_t)(lane & 15) * ld + ((lane >> 4) << 3);
  v16bf b;
  ((float4*)&b)[0] = *(const float4*)(p);
  ((float4*)&b)[1] = *(const float4*)(p + 16);
  return b;
}

__device__ __forceinline__ v8f wmma_bf16(v16bf a, v16bf b, v8f c) {
  // (neg_a, A, neg_b, B, c_mod, C, reuse_a, reuse_b)
  return __builtin_amdgcn_wmma_f32_16x16x32_bf16(false, a, false, b, (short)0, c,
                                                 false, false);
}

__device__ __forceinline__ v8f vzero8() {
  v8f z = {0.f, 0.f, 0.f, 0.f, 0.f, 0.f, 0.f, 0.f};
  return z;
}

__device__ __forceinline__ float elu1(float x) {
  return x > 0.f ? x + 1.f : __expf(x);  // elu(x)+1
}

// --- CDNA5 async global->LDS (ASYNCcnt-tracked, bypasses VGPRs) ---
// VDST = per-lane LDS byte offset (low 32 bits of the generic shared-memory
// address ARE the LDS address per the flat-aperture mapping); VADDR = 64-bit
// global address; "off" = no SADDR (GV mode).
__device__ __forceinline__ void async_b128(void* lds_ptr, const bf16* g) {
  uint32_t lds_off = (uint32_t)(uintptr_t)lds_ptr;
  asm volatile("global_load_async_to_lds_b128 %0, %1, off"
               :
               : "v"(lds_off), "v"(g)
               : "memory");
}
__device__ __forceinline__ void wait_async_le3() {
#if __has_builtin(__builtin_amdgcn_s_wait_asynccnt)
  __builtin_amdgcn_s_wait_asynccnt(3);
#else
  asm volatile("s_wait_asynccnt 0x3" ::: "memory");
#endif
}
__device__ __forceinline__ void wait_async_0() {
#if __has_builtin(__builtin_amdgcn_s_wait_asynccnt)
  __builtin_amdgcn_s_wait_asynccnt(0);
#else
  asm volatile("s_wait_asynccnt 0x0" ::: "memory");
#endif
}

// ---------------- conversion kernels ----------------

__global__ __launch_bounds__(256) void cvt_f32_bf16(const float* __restrict__ in,
                                                    bf16* __restrict__ out, int n) {
  for (int i = blockIdx.x * blockDim.x + threadIdx.x; i < n;
       i += gridDim.x * blockDim.x)
    out[i] = (bf16)in[i];
}

// proj_matrix [H][D][F] f32  ->  pmt [H][F][D] bf16 (row f contiguous in d)
__global__ __launch_bounds__(256) void cvt_pm(const float* __restrict__ pm,
                                              bf16* __restrict__ pmt) {
  int i = blockIdx.x * blockDim.x + threadIdx.x;
  if (i >= H_NUM * F_DIM * D_HEAD) return;
  int d = i & (D_HEAD - 1);
  int f = (i >> 6) & (F_DIM - 1);
  int h = i >> 13;
  pmt[i] = (bf16)pm[((size_t)h * D_HEAD + d) * F_DIM + f];
}

// ---------------- generic GEMM-NT: C[M,N] = A[M,K] * W[N,K]^T ----------------
// block tile 64x128, 8 waves (2 M x 4 N), wave tile 32x32, K step 32.
// LDS tiles are double-buffered and filled with async global->LDS loads:
// prefetch tile s+1 while WMMAs consume tile s; s_wait_asynccnt 3 means the
// current tile's 3 per-thread async ops have landed (in-order completion).
template <bool BF16_OUT, bool BIAS>
__global__ __launch_bounds__(256) void gemm_nt(const bf16* __restrict__ A,
                                               const bf16* __restrict__ W,
                                               const float* __restrict__ bias,
                                               void* __restrict__ Cout, int M, int N,
                                               int K) {
  __shared__ __align__(16) bf16 lA[2][64 * 40];    // pad rows to 40 elems (80B)
  __shared__ __align__(16) bf16 lB[2][128 * 40];
  const int tid = threadIdx.x;
  const int lane = tid & 31;
  const int w = tid >> 5;
  const int wm = w >> 2;   // 0..1
  const int wn = w & 3;    // 0..3
  const int m0 = blockIdx.x * 64;
  const int n0 = blockIdx.y * 128;

  // per-thread staging assignment (16B-aligned chunks)
  const int arow = tid >> 2;             // 0..63
  const int akc = (tid & 3) << 3;        // 0,8,16,24
  const int brow = tid >> 1;             // 0..127
  const int bkc = (tid & 1) << 4;        // 0,16
  const bf16* Ag = &A[(size_t)(m0 + arow) * K + akc];
  const bf16* Bg = &W[(size_t)(n0 + brow) * K + bkc];

  v8f acc[2][2];
  for (int i = 0; i < 2; ++i)
    for (int j = 0; j < 2; ++j) acc[i][j] = vzero8();

  // prologue: async-fill buffer 0 with k-step 0
  async_b128(&lA[0][arow * 40 + akc], Ag);
  async_b128(&lB[0][brow * 40 + bkc], Bg);
  async_b128(&lB[0][brow * 40 + bkc + 8], Bg + 8);

  const int nsteps = K >> 5;
  for (int s = 0; s < nsteps; ++s) {
    const int buf = s & 1;
    if (s + 1 < nsteps) {
      const int k1 = (s + 1) << 5;
      async_b128(&lA[buf ^ 1][arow * 40 + akc], Ag + k1);
      async_b128(&lB[buf ^ 1][brow * 40 + bkc], Bg + k1);
      async_b128(&lB[buf ^ 1][brow * 40 + bkc + 8], Bg + k1 + 8);
      wait_async_le3();  // the 3 ops of step s are done; s+1 still in flight
    } else {
      wait_async_0();
    }
    __syncthreads();  // all waves' tile-s data visible in LDS

    v16bf af[2], bfr[2];
    for (int i = 0; i < 2; ++i)
      af[i] = load_frag_a(&lA[buf][(wm * 32 + i * 16) * 40], 40, lane);
    for (int j = 0; j < 2; ++j)
      bfr[j] = load_frag_b(&lB[buf][(wn * 32 + j * 16) * 40], 40, lane);
    for (int i = 0; i < 2; ++i)
      for (int j = 0; j < 2; ++j) acc[i][j] = wmma_bf16(af[i], bfr[j], acc[i][j]);
    __syncthreads();  // all reads of buf done; next iter may async-overwrite it
  }

  const int col = lane & 15;
  const int rb = (lane >> 4) << 3;
  for (int i = 0; i < 2; ++i)
    for (int j = 0; j < 2; ++j) {
      int n = n0 + wn * 32 + j * 16 + col;
      float bv = BIAS ? bias[n] : 0.f;
      for (int r = 0; r < 8; ++r) {
        int m = m0 + wm * 32 + i * 16 + rb + r;
        float v = acc[i][j][r] + bv;
        if (BF16_OUT)
          ((bf16*)Cout)[(size_t)m * N + n] = (bf16)v;
        else
          ((float*)Cout)[(size_t)m * N + n] = v;
      }
    }
}

// ---------------- favor projection for Q: Q[m, h*128+f] = elu1(q_h @ pmT_h^T) ----
// grid (BN/64, H), block 256; wave tile 32x32, K = 64 (2 steps), direct-global.
__global__ __launch_bounds__(256) void favor_proj(const bf16* __restrict__ QKV,
                                                  const bf16* __restrict__ PMT,
                                                  bf16* __restrict__ Qout) {
  const int h = blockIdx.y;
  const int m0 = blockIdx.x * 64;
  const int tid = threadIdx.x, lane = tid & 31, w = tid >> 5;
  const int wm = w >> 2, wn = w & 3;
  const bf16* Abase = QKV + (size_t)m0 * C3_DIM + h * D_HEAD;  // q part, cols h*64..
  const bf16* Bbase = PMT + (size_t)h * F_DIM * D_HEAD;

  v8f acc[2][2];
  for (int i = 0; i < 2; ++i)
    for (int j = 0; j < 2; ++j) acc[i][j] = vzero8();

  for (int k0 = 0; k0 < D_HEAD; k0 += 32) {
    v16bf af[2], bfr[2];
    for (int i = 0; i < 2; ++i)
      af[i] = load_frag_a(Abase + (size_t)(wm * 32 + i * 16) * C3_DIM + k0, C3_DIM, lane);
    for (int j = 0; j < 2; ++j)
      bfr[j] = load_frag_b(Bbase + (size_t)(wn * 32 + j * 16) * D_HEAD + k0, D_HEAD, lane);
    for (int i = 0; i < 2; ++i)
      for (int j = 0; j < 2; ++j) acc[i][j] = wmma_bf16(af[i], bfr[j], acc[i][j]);
  }

  const int col = lane & 15;
  const int rb = (lane >> 4) << 3;
  for (int i = 0; i < 2; ++i)
    for (int j = 0; j < 2; ++j) {
      int f = wn * 32 + j * 16 + col;
      for (int r = 0; r < 8; ++r) {
        int m = m0 + wm * 32 + i * 16 + rb + r;
        Qout[(size_t)m * (H_NUM * F_DIM) + h * F_DIM + f] = (bf16)elu1(acc[i][j][r]);
      }
    }
}

// ------------- fused favor(K) + KV partial:  KVpart[chunk][h] += K_h^T V_h -------
// grid (KV_CHUNKS, H). Each block: 512 rows in 4 sub-chunks of 128.
// Phase 1: K_favor tile [128,128] -> LDS transposed KT[f][m].
// Phase 0: V tile -> LDS transposed VT[d][m].
// Phase 2: WMMA with K-dim = sequence index m (4 k-steps of 32 per sub-chunk).
__global__ __launch_bounds__(256) void kv_partial(const bf16* __restrict__ QKV,
                                                  const bf16* __restrict__ PMT,
                                                  float* __restrict__ KVpart) {
  __shared__ __align__(16) bf16 KT[F_DIM * 136];   // [f][m], stride 136 (272B)
  __shared__ __align__(16) bf16 VT[D_HEAD * 136];  // [d][m]
  const int h = blockIdx.y;
  const int chunk = blockIdx.x;
  const int tid = threadIdx.x, lane = tid & 31, w = tid >> 5;
  const int col = lane & 15;
  const int rb = (lane >> 4) << 3;
  const int pf = w >> 1;  // 0..3 : 32-wide f block of KV output
  const int pd = w & 1;   // 0..1 : 32-wide d block
  const bf16* Bpm = PMT + (size_t)h * F_DIM * D_HEAD;

  v8f kv[2][2];
  for (int i = 0; i < 2; ++i)
    for (int j = 0; j < 2; ++j) kv[i][j] = vzero8();

  for (int sc = 0; sc < 4; ++sc) {
    const int m0 = (chunk * 4 + sc) * 128;
    __syncthreads();  // previous phase-2 reads done before LDS overwrite

    // phase 0: V[m, h*64+d] -> VT[d][m]
    for (int idx = tid; idx < 128 * D_HEAD; idx += 256) {
      int m = idx >> 6, d = idx & (D_HEAD - 1);
      VT[d * 136 + m] =
          QKV[(size_t)(m0 + m) * C3_DIM + 2 * C_DIM + h * D_HEAD + d];
    }

    // phase 1: K_favor = elu1(k_h[128,64] @ pmT_h^T) -> KT[f][m] (transposed)
    const bf16* Ak = QKV + (size_t)m0 * C3_DIM + C_DIM + h * D_HEAD;
    for (int t = 0; t < 2; ++t) {
      int id = w * 2 + t;          // 16 tiles of 16x16
      int ti = id >> 2;            // m group
      int tj = id & 3;             // f group
      v8f a = vzero8();
      for (int k0 = 0; k0 < D_HEAD; k0 += 32) {
        v16bf fa = load_frag_a(Ak + (size_t)(ti * 16) * C3_DIM + k0, C3_DIM, lane);
        v16bf fb = load_frag_b(Bpm + (size_t)(tj * 16) * D_HEAD + k0, D_HEAD, lane);
        a = wmma_bf16(fa, fb, a);
      }
      int f = tj * 16 + col;
      for (int r = 0; r < 8; ++r) {
        int m = ti * 16 + rb + r;
        KT[f * 136 + m] = (bf16)elu1(a[r]);
      }
    }
    __syncthreads();

    // phase 2: KV[f,d] += KT^T-view: A[f, m] = KT, B[m, d] via VT[d][m]
    for (int k0 = 0; k0 < 128; k0 += 32) {
      v16bf fa[2], fb[2];
      for (int i = 0; i < 2; ++i)
        fa[i] = load_frag_a(&KT[(pf * 32 + i * 16) * 136 + k0], 136, lane);
      for (int j = 0; j < 2; ++j)
        fb[j] = load_frag_b(&VT[(pd * 32 + j * 16) * 136 + k0], 136, lane);
      for (int i = 0; i < 2; ++i)
        for (int j = 0; j < 2; ++j) kv[i][j] = wmma_bf16(fa[i], fb[j], kv[i][j]);
    }
  }

  float* out = KVpart + ((size_t)chunk * H_NUM + h) * F_DIM * D_HEAD;
  for (int i = 0; i < 2; ++i)
    for (int j = 0; j < 2; ++j) {
      int d = pd * 32 + j * 16 + col;
      for (int r = 0; r < 8; ++r) {
        int f = pf * 32 + i * 16 + rb + r;
        out[(size_t)f * D_HEAD + d] = kv[i][j][r];
      }
    }
}

// deterministic reduce over chunks + transpose: KVt[h][d][f] bf16
__global__ __launch_bounds__(256) void kv_finalize(const float* __restrict__ KVpart,
                                                   bf16* __restrict__ KVt) {
  int i = blockIdx.x * blockDim.x + threadIdx.x;
  if (i >= H_NUM * F_DIM * D_HEAD) return;
  int d = i & (D_HEAD - 1);
  int f = (i >> 6) & (F_DIM - 1);
  int h = i >> 13;
  float s = 0.f;
  for (int c = 0; c < KV_CHUNKS; ++c)
    s += KVpart[(((size_t)c * H_NUM + h) * F_DIM + f) * D_HEAD + d];
  KVt[((size_t)h * D_HEAD + d) * F_DIM + f] = (bf16)s;
}

// ---------- O1[m, h*64+d] = (Q_h @ KV_h) * Z ; Z = 1/rowsum(Q_h @ KV_h) ----------
// grid (BN/128, H); 8 waves, each wave = one 16x64 strip (4 N-tiles), K = 128.
__global__ __launch_bounds__(256) void apply_kv(const bf16* __restrict__ Q,
                                                const bf16* __restrict__ KVt,
                                                bf16* __restrict__ O1) {
  const int h = blockIdx.y;
  const int m0 = blockIdx.x * 128;
  const int tid = threadIdx.x, lane = tid & 31, w = tid >> 5;
  const bf16* Abase = Q + (size_t)(m0 + w * 16) * (H_NUM * F_DIM) + h * F_DIM;
  const bf16* Bbase = KVt + (size_t)h * D_HEAD * F_DIM;

  v8f acc[4];
  for (int j = 0; j < 4; ++j) acc[j] = vzero8();

  for (int k0 = 0; k0 < F_DIM; k0 += 32) {
    v16bf a = load_frag_a(Abase + k0, H_NUM * F_DIM, lane);
    for (int j = 0; j < 4; ++j) {
      v16bf b = load_frag_b(Bbase + (size_t)(j * 16) * F_DIM + k0, F_DIM, lane);
      acc[j] = wmma_bf16(a, b, acc[j]);
    }
  }

  // per-row normalizer: reduce over 4 tiles and across the 16-lane half
  for (int r = 0; r < 8; ++r) {
    float s = acc[0][r] + acc[1][r] + acc[2][r] + acc[3][r];
    s += __shfl_xor(s, 1);
    s += __shfl_xor(s, 2);
    s += __shfl_xor(s, 4);
    s += __shfl_xor(s, 8);
    float zinv = 1.f / s;
    for (int j = 0; j < 4; ++j) acc[j][r] *= zinv;
  }

  const int col = lane & 15;
  const int rb = (lane >> 4) << 3;
  for (int j = 0; j < 4; ++j)
    for (int r = 0; r < 8; ++r) {
      int m = m0 + w * 16 + rb + r;
      O1[(size_t)m * C_DIM + h * D_HEAD + j * 16 + col] = (bf16)acc[j][r];
    }
}

// ---------------- workspace layout (bytes, 256-aligned) ----------------
#define OFF_XBF    ((size_t)0)                       // 33,554,432
#define OFF_WQKV   ((size_t)33554432)                //  1,572,864
#define OFF_WPROJ  ((size_t)35127296)                //    524,288
#define OFF_PMT    ((size_t)35651584)                //    131,072
#define OFF_QKV    ((size_t)35782656)                // 100,663,296
#define OFF_Q      ((size_t)136445952)               //  67,108,864
#define OFF_KVPART ((size_t)203554816)               //  16,777,216
#define OFF_KVT    ((size_t)220332032)               //    131,072
#define OFF_O1     ((size_t)220463104)               //  33,554,432
#define WS_NEEDED  ((size_t)254017536)

extern "C" void kernel_launch(void* const* d_in, const int* in_sizes, int n_in,
                              void* d_out, int out_size, void* d_ws, size_t ws_size,
                              hipStream_t stream) {
  if (ws_size < WS_NEEDED) return;
  const float* x      = (const float*)d_in[0];   // [8,4096,512]
  const float* w_qkv  = (const float*)d_in[1];   // [1536,512]
  const float* w_proj = (const float*)d_in[2];   // [512,512]
  const float* b_proj = (const float*)d_in[3];   // [512]
  const float* pm     = (const float*)d_in[4];   // [8,64,128]

  char* ws = (char*)d_ws;
  bf16* xbf    = (bf16*)(ws + OFF_XBF);
  bf16* wqkvb  = (bf16*)(ws + OFF_WQKV);
  bf16* wprojb = (bf16*)(ws + OFF_WPROJ);
  bf16* pmt    = (bf16*)(ws + OFF_PMT);
  bf16* qkvb   = (bf16*)(ws + OFF_QKV);
  bf16* qb     = (bf16*)(ws + OFF_Q);
  float* kvp   = (float*)(ws + OFF_KVPART);
  bf16* kvt    = (bf16*)(ws + OFF_KVT);
  bf16* o1b    = (bf16*)(ws + OFF_O1);

  cvt_f32_bf16<<<4096, 256, 0, stream>>>(x, xbf, BN_TOT * C_DIM);
  cvt_f32_bf16<<<768, 256, 0, stream>>>(w_qkv, wqkvb, C3_DIM * C_DIM);
  cvt_f32_bf16<<<256, 256, 0, stream>>>(w_proj, wprojb, C_DIM * C_DIM);
  cvt_pm<<<256, 256, 0, stream>>>(pm, pmt);

  // qkv[BN,1536] = x @ w_qkv^T  (bf16 out)
  gemm_nt<true, false><<<dim3(BN_TOT / 64, C3_DIM / 128), 256, 0, stream>>>(
      xbf, wqkvb, nullptr, qkvb, BN_TOT, C3_DIM, C_DIM);

  // Q = elu(q @ pm)+1  -> [BN, H*F] bf16
  favor_proj<<<dim3(BN_TOT / 64, H_NUM), 256, 0, stream>>>(qkvb, pmt, qb);

  // KV partials (fused favor(K) + K^T V), then deterministic reduce+transpose
  kv_partial<<<dim3(KV_CHUNKS, H_NUM), 256, 0, stream>>>(qkvb, pmt, kvp);
  kv_finalize<<<256, 256, 0, stream>>>(kvp, kvt);

  // O1 = (Q @ KV) * Z   -> [BN, 512] bf16
  apply_kv<<<dim3(BN_TOT / 128, H_NUM), 256, 0, stream>>>(qb, kvt, o1b);

  // out = O1 @ w_proj^T + b_proj  (f32 out)
  gemm_nt<false, true><<<dim3(BN_TOT / 64, C_DIM / 128), 256, 0, stream>>>(
      o1b, wprojb, b_proj, (float*)d_out, BN_TOT, C_DIM, C_DIM);
}
